// PointerNetwork_15075335209262
// MI455X (gfx1250) — compile-verified, matrix operand order checked by reading, not run on hardware
//
#include <hip/hip_runtime.h>
#include <hip/hip_bf16.h>

typedef __attribute__((ext_vector_type(16))) __bf16 v16bf;
typedef __attribute__((ext_vector_type(8)))  float  v8f;

#define NEG_INF (-1e9f)

static constexpr int Bsz  = 64;
static constexpr int Nseq = 2048;
static constexpr int Dd   = 256;   // == H == P == K for every matmul here
static constexpr int H3   = 768;

// ---------------------------------------------------------------------------
// Pack a weight matrix (f32) into WMMA B-fragment order, bf16.
// Logical B is [256, Ncol] (K x N):
//   transpose==0: W stored [256, Ncol], B(k,n) = W[k*Ncol + n]
//   transpose==1: W stored [Ncol, 256], B(k,n) = W[n*256 + k]   (for W_ih/W_hh)
// Output layout: element ((colTile*8 + kc)*32 + lane)*16 + s, so each lane's
// 16 bf16 fragment elements are 32 contiguous, 32B-aligned bytes.
// ---------------------------------------------------------------------------
__global__ void pack_weight_bf16(const float* __restrict__ W, __bf16* __restrict__ out,
                                 int Ncol, int transpose) {
    int idx = blockIdx.x * blockDim.x + threadIdx.x;   // over 256*Ncol
    if (idx >= 256 * Ncol) return;
    int s    = idx & 15;
    int lane = (idx >> 4) & 31;
    int kc   = (idx >> 9) & 7;
    int ct   = idx >> 12;
    int n    = ct * 16 + (lane & 15);
    // 16-bit B layout: slot s -> K = 16*(s/8) + 8*(lane/16) + (s%8)
    int kl   = ((s >> 3) << 4) + ((lane >> 4) << 3) + (s & 7);
    int k    = kc * 32 + kl;
    float v  = transpose ? W[(size_t)n * 256 + k] : W[(size_t)k * Ncol + n];
    out[idx] = (__bf16)v;
}

__global__ void copy_f32(const float* __restrict__ in, float* __restrict__ out, int n) {
    int i = blockIdx.x * blockDim.x + threadIdx.x;
    if (i < n) out[i] = in[i];
}

// ---------------------------------------------------------------------------
// WMMA GEMM: C[M, Ncol] = A[M, 256] (f32, bf16-converted on the fly) x
//            Wp (packed bf16 B fragments, L2-resident).
// Block = 256 threads = 8 waves; block covers 16 rows x 256 cols
// (each wave owns 2 adjacent 16x16 col tiles -> A fragment reused twice).
// grid.x = M/16, grid.y = Ncol/256.  A is read from HBM exactly once.
// ---------------------------------------------------------------------------
__global__ __launch_bounds__(256) void gemm_wmma_bf16(
    const float* __restrict__ A, const __bf16* __restrict__ Wp,
    float* __restrict__ C, int M, int Ncol) {
    __shared__ float tileA[16 * 256];   // 16 KB

    const int tid  = threadIdx.x;
    const int row0 = blockIdx.x * 16;

    // One contiguous, fully coalesced 16 KB copy (tile rows are contiguous).
#pragma unroll
    for (int i = 0; i < 16; ++i) {
        int idx = tid + i * 256;
        tileA[idx] = A[(size_t)row0 * 256 + idx];
    }
    __syncthreads();

    const int wave = tid >> 5;
    const int lane = tid & 31;
    const int nloc = lane & 15;
    const int hi   = lane >> 4;
    const int kgrp = hi * 8;
    const int ct0  = blockIdx.y * 16 + wave * 2;   // first of 2 col tiles

    v8f acc0 = {}, acc1 = {};
#pragma unroll
    for (int kc = 0; kc < 8; ++kc) {
        const int k0 = kc * 32;
        // A fragment: two 8-float contiguous runs (32B-aligned) from LDS.
        const float4* ap0 = (const float4*)&tileA[nloc * 256 + k0 + kgrp];
        const float4* ap1 = (const float4*)&tileA[nloc * 256 + k0 + 16 + kgrp];
        float4 f0 = ap0[0], f1 = ap0[1], f2 = ap1[0], f3 = ap1[1];
        v16bf a;
        a[0]=(__bf16)f0.x;  a[1]=(__bf16)f0.y;  a[2]=(__bf16)f0.z;  a[3]=(__bf16)f0.w;
        a[4]=(__bf16)f1.x;  a[5]=(__bf16)f1.y;  a[6]=(__bf16)f1.z;  a[7]=(__bf16)f1.w;
        a[8]=(__bf16)f2.x;  a[9]=(__bf16)f2.y;  a[10]=(__bf16)f2.z; a[11]=(__bf16)f2.w;
        a[12]=(__bf16)f3.x; a[13]=(__bf16)f3.y; a[14]=(__bf16)f3.z; a[15]=(__bf16)f3.w;

        // B fragments: one 32B contiguous load per tile per lane.
        v16bf b0 = *(const v16bf*)(Wp + ((size_t)(ct0 * 8 + kc) * 32 + lane) * 16);
        v16bf b1 = *(const v16bf*)(Wp + ((size_t)((ct0 + 1) * 8 + kc) * 32 + lane) * 16);

        acc0 = __builtin_amdgcn_wmma_f32_16x16x32_bf16(false, a, false, b0,
                                                       (short)0, acc0, false, false);
        acc1 = __builtin_amdgcn_wmma_f32_16x16x32_bf16(false, a, false, b1,
                                                       (short)0, acc1, false, false);
    }

    // D layout: lanes 0-15 -> M=r, lanes 16-31 -> M=8+r; N = nloc.
#pragma unroll
    for (int r = 0; r < 8; ++r) {
        int row = row0 + hi * 8 + r;
        C[(size_t)row * Ncol + ct0 * 16 + nloc]       = acc0[r];
        C[(size_t)row * Ncol + (ct0 + 1) * 16 + nloc] = acc1[r];
    }
}

// ---------------------------------------------------------------------------
// scores[b,n] = v . tanh(seq_proj[b,n,:] + hWh[b,:]) + mask_bias
// ---------------------------------------------------------------------------
__global__ __launch_bounds__(256) void attn_scores(
    const float* __restrict__ seq_proj, const float* __restrict__ hWh,
    const float* __restrict__ v_att, const float* __restrict__ mask,
    float* __restrict__ scores) {
    __shared__ float sv[256], sh[256];
    const int b = blockIdx.y;
    const int tid = threadIdx.x;
    sv[tid] = v_att[tid];
    sh[tid] = hWh[b * 256 + tid];
    __syncthreads();

    const int wave = tid >> 5, lane = tid & 31;
    const int n = blockIdx.x * 8 + wave;
    const float* row = seq_proj + ((size_t)b * Nseq + n) * 256;

    float acc = 0.f;
#pragma unroll
    for (int j = 0; j < 8; ++j) {
        int p = lane + j * 32;
        acc += sv[p] * tanhf(row[p] + sh[p]);
    }
    for (int off = 16; off > 0; off >>= 1) acc += __shfl_down(acc, off, 32);
    if (lane == 0) {
        float mb = (mask[(size_t)b * Nseq + n] > 0.f) ? 0.f : NEG_INF;
        scores[(size_t)b * Nseq + n] = acc + mb;
    }
}

// ---------------------------------------------------------------------------
// Row softmax over N=2048 -> output tensor [B, T, N] at step t.
// ---------------------------------------------------------------------------
__global__ __launch_bounds__(256) void softmax_row(
    const float* __restrict__ scores, float* __restrict__ out, int t, int T) {
    __shared__ float red[256];
    const int b = blockIdx.x, tid = threadIdx.x;
    const float* s = scores + (size_t)b * Nseq;

    float mx = -3.4e38f;
    for (int n = tid; n < Nseq; n += 256) mx = fmaxf(mx, s[n]);
    red[tid] = mx; __syncthreads();
    for (int w = 128; w > 0; w >>= 1) { if (tid < w) red[tid] = fmaxf(red[tid], red[tid + w]); __syncthreads(); }
    mx = red[0]; __syncthreads();

    float sum = 0.f;
    for (int n = tid; n < Nseq; n += 256) sum += __expf(s[n] - mx);
    red[tid] = sum; __syncthreads();
    for (int w = 128; w > 0; w >>= 1) { if (tid < w) red[tid] += red[tid + w]; __syncthreads(); }
    const float inv = 1.f / red[0];

    float* o = out + ((size_t)b * T + t) * Nseq;
    for (int n = tid; n < Nseq; n += 256) o[n] = __expf(s[n] - mx) * inv;
}

// ---------------------------------------------------------------------------
// vec[b,:] = sum_n w[b,n] * sequence[b,n,:]
// ---------------------------------------------------------------------------
__global__ __launch_bounds__(256) void weighted_sum(
    const float* __restrict__ w, const float* __restrict__ seq,
    float* __restrict__ vec, int t, int T) {
    __shared__ float sw[256];
    const int b = blockIdx.x, tid = threadIdx.x;
    const float* wb = w + ((size_t)b * T + t) * Nseq;
    const float* sb = seq + (size_t)b * Nseq * Dd;

    float acc = 0.f;
    for (int n0 = 0; n0 < Nseq; n0 += 256) {
        __syncthreads();
        sw[tid] = wb[n0 + tid];
        __syncthreads();
        if (n0 + 256 < Nseq)  // global_prefetch_b8 on the next streamed chunk
            __builtin_prefetch(&sb[(size_t)(n0 + 256) * Dd + tid], 0, 0);
        for (int j = 0; j < 256; ++j)
            acc += sw[j] * sb[(size_t)(n0 + j) * Dd + tid];
    }
    vec[(size_t)b * Dd + tid] = acc;
}

// ---------------------------------------------------------------------------
// GRU gate math; updates h in place.
// ---------------------------------------------------------------------------
__global__ __launch_bounds__(256) void gru_gates(
    const float* __restrict__ gx, const float* __restrict__ gh,
    const float* __restrict__ b_ih, const float* __restrict__ b_hh,
    float* __restrict__ h) {
    const int i = blockIdx.x * 256 + threadIdx.x;   // over B*H
    const int b = i >> 8, j = i & 255;
    const size_t g0 = (size_t)b * H3;

    float xr = gx[g0 + j]       + b_ih[j],       hr = gh[g0 + j]       + b_hh[j];
    float xz = gx[g0 + 256 + j] + b_ih[256 + j], hz = gh[g0 + 256 + j] + b_hh[256 + j];
    float xn = gx[g0 + 512 + j] + b_ih[512 + j], hn = gh[g0 + 512 + j] + b_hh[512 + j];

    float r  = 1.f / (1.f + __expf(-(xr + hr)));
    float z  = 1.f / (1.f + __expf(-(xz + hz)));
    float nn = tanhf(xn + r * hn);
    float hv = h[(size_t)b * 256 + j];
    h[(size_t)b * 256 + j] = (1.f - z) * nn + z * hv;
}

// ---------------------------------------------------------------------------
extern "C" void kernel_launch(void* const* d_in, const int* in_sizes, int n_in,
                              void* d_out, int out_size, void* d_ws, size_t ws_size,
                              hipStream_t stream) {
    const float* sequence = (const float*)d_in[0];   // [B, N, D]
    const float* hidden_t = (const float*)d_in[1];   // [B, H]
    const float* seq_mask = (const float*)d_in[2];   // [B, N]
    const float* W_seq    = (const float*)d_in[4];   // [D, P]
    const float* W_h      = (const float*)d_in[5];   // [H, P]
    const float* v_att    = (const float*)d_in[6];   // [P]
    const float* W_ih     = (const float*)d_in[7];   // [3H, D]
    const float* W_hh     = (const float*)d_in[8];   // [3H, H]
    const float* b_ih     = (const float*)d_in[9];   // [3H]
    const float* b_hh     = (const float*)d_in[10];  // [3H]
    float* out = (float*)d_out;                      // [B, T, N]
    (void)in_sizes; (void)n_in; (void)ws_size;

    const int T = out_size / (Bsz * Nseq);           // == num_steps (2)

    // Workspace carve-up (256B aligned).
    char* ws = (char*)d_ws;
    size_t off = 0;
    auto carve = [&](size_t bytes) { size_t o = off; off += (bytes + 255) & ~(size_t)255; return o; };
    float*  seq_proj = (float*)(ws + carve((size_t)Bsz * Nseq * Dd * 4));  // 128 MB (L2-resident)
    __bf16* Wseq_p   = (__bf16*)(ws + carve(256 * 256 * 2));
    __bf16* Wh_p     = (__bf16*)(ws + carve(256 * 256 * 2));
    __bf16* Wih_p    = (__bf16*)(ws + carve(256 * H3 * 2));
    __bf16* Whh_p    = (__bf16*)(ws + carve(256 * H3 * 2));
    float*  hWh      = (float*)(ws + carve((size_t)Bsz * 256 * 4));
    float*  scores   = (float*)(ws + carve((size_t)Bsz * Nseq * 4));
    float*  vecb     = (float*)(ws + carve((size_t)Bsz * Dd * 4));
    float*  gx       = (float*)(ws + carve((size_t)Bsz * H3 * 4));
    float*  gh       = (float*)(ws + carve((size_t)Bsz * H3 * 4));
    float*  h        = (float*)(ws + carve((size_t)Bsz * 256 * 4));

    // One-time weight packing + hidden-state copy.
    pack_weight_bf16<<<256, 256, 0, stream>>>(W_seq, Wseq_p, 256, 0);
    pack_weight_bf16<<<256, 256, 0, stream>>>(W_h,   Wh_p,   256, 0);
    pack_weight_bf16<<<768, 256, 0, stream>>>(W_ih,  Wih_p,  H3,  1);
    pack_weight_bf16<<<768, 256, 0, stream>>>(W_hh,  Whh_p,  H3,  1);
    copy_f32<<<64, 256, 0, stream>>>(hidden_t, h, Bsz * 256);

    // Big GEMM: seq_proj[131072, 256] = sequence x W_seq (single HBM pass on A)
    gemm_wmma_bf16<<<dim3(Bsz * Nseq / 16, 1), 256, 0, stream>>>(
        sequence, Wseq_p, seq_proj, Bsz * Nseq, 256);

    for (int t = 0; t < T; ++t) {
        // hWh[64, 256] = h x W_h
        gemm_wmma_bf16<<<dim3(4, 1), 256, 0, stream>>>(h, Wh_p, hWh, Bsz, 256);
        // scores (streams seq_proj; hits L2)
        attn_scores<<<dim3(Nseq / 8, Bsz), 256, 0, stream>>>(seq_proj, hWh, v_att, seq_mask, scores);
        // softmax -> out[:, t, :]
        softmax_row<<<Bsz, 256, 0, stream>>>(scores, out, t, T);
        // vec = w x sequence
        weighted_sum<<<Bsz, 256, 0, stream>>>(out, sequence, vecb, t, T);
        // GRU matmuls as WMMA GEMMs: gx = vec x W_ih^T, gh = h x W_hh^T
        gemm_wmma_bf16<<<dim3(4, H3 / 256), 256, 0, stream>>>(vecb, Wih_p, gx, Bsz, H3);
        gemm_wmma_bf16<<<dim3(4, H3 / 256), 256, 0, stream>>>(h,    Whh_p, gh, Bsz, H3);
        // gates + state update
        gru_gates<<<Bsz * 256 / 256, 256, 0, stream>>>(gx, gh, b_ih, b_hh, h);
    }
}
